// LinearAttention_37048387895748
// MI455X (gfx1250) — matrix operand order repaired and checked by reading, not compiled, not run on hardware
//
#include <hip/hip_runtime.h>
#include <hip/hip_bf16.h>
#include <stdint.h>

// Problem constants (reference: B=8, S=4096, DIM=1024, H=16, HD=64)
#define B_    8
#define S_    4096
#define DIM_  1024
#define H_    16
#define HD_   64
#define SCALE_ 0.125f   // HD^-0.5

typedef unsigned short u16;
typedef __attribute__((ext_vector_type(16))) __bf16 v16bf;
typedef __attribute__((ext_vector_type(8)))  float  v8f;

// bf16 <-> f32 via raw bits (round-to-nearest-even); avoids __bf16 scalar arith.
__device__ __forceinline__ u16 f2bf(float f) {
    uint32_t u = __float_as_uint(f);
    uint32_t r = u + 0x7FFFu + ((u >> 16) & 1u);
    return (u16)(r >> 16);
}
__device__ __forceinline__ float bf2f(u16 h) {
    return __uint_as_float(((uint32_t)h) << 16);
}

// One wave's 16x32 (A) or 32x16 (B) bf16 WMMA operand: 8 VGPRs = 16 bf16/lane.
union Frag16 { v16bf v; uint4 q[2]; };

__device__ __forceinline__ void loadA(Frag16& f, const u16* row, int k0, int half) {
    f.q[0] = *(const uint4*)(row + k0 + half * 16);
    f.q[1] = *(const uint4*)(row + k0 + half * 16 + 8);
}
__device__ __forceinline__ void loadB(Frag16& f, const u16* row, int k0, int half) {
    f.q[0] = *(const uint4*)(row + k0 + half * 16);
    f.q[1] = *(const uint4*)(row + k0 + half * 16 + 8);
}
// B fragment from an LDS tile row (row stride BSTRIDE elems, tile is 32 K wide).
__device__ __forceinline__ void loadB_lds(Frag16& f, const u16* ldsRow, int half) {
    f.q[0] = *(const uint4*)(ldsRow + half * 16);
    f.q[1] = *(const uint4*)(ldsRow + half * 16 + 8);
}
__device__ __forceinline__ v8f wmma_bf16(const Frag16& a, const Frag16& b, v8f c) {
    return __builtin_amdgcn_wmma_f32_16x16x32_bf16(false, a.v, false, b.v,
                                                   (short)0, c, false, false);
}

// CDNA5 async copy global->LDS (per-lane 16B), tracked by ASYNCcnt.
__device__ __forceinline__ void async_g2l_b128(uint32_t ldsByteAddr, const void* gaddr) {
    asm volatile("global_load_async_to_lds_b128 %0, %1, off"
                 :: "v"(ldsByteAddr), "v"((uint64_t)(uintptr_t)gaddr) : "memory");
}
__device__ __forceinline__ void wait_async0() {
    asm volatile("s_wait_asynccnt 0" ::: "memory");
}

#define BSTRIDE 40   // padded LDS row stride (bf16 elems): 80B = 20 banks -> 16-lane conflict-free

// ---------------------------------------------------------------------------
// Register-blocked bf16 GEMM: C[M,N] = A[M,K] x BT[N,K] + bias.
// Wave tile: 32x64 (2x4 WMMA accumulators). Block: 8 waves 4(M) x 2(N) ->
// 128x128 macro-tile. The 128x32 B tile is staged in LDS via double-buffered
// GLOBAL_LOAD_ASYNC_TO_LDS_B128 (exactly the unique 8KB per k-step, killing
// the 4x B overfetch of direct loads); A streams from global/L2.
// Grid: (N/128, M/128).
// ---------------------------------------------------------------------------
__global__ __launch_bounds__(256) void gemm_bf16_wmma(
    const u16* __restrict__ A, const u16* __restrict__ BT,
    const float* __restrict__ bias, void* __restrict__ out,
    int N, int K, int outF32)
{
    __shared__ u16 bsmem[2][128 * BSTRIDE];   // 2 x 10240B

    const int lane = threadIdx.x & 31;
    const int w    = threadIdx.x >> 5;
    const int half = lane >> 4;
    const int ln   = lane & 15;
    const int wm = w & 3, wn = w >> 2;
    const int nBlock = blockIdx.x * 128;
    const int m0 = blockIdx.y * 128 + wm * 32;
    const int n0 = nBlock + wn * 64;

    const u16* aRow0 = A + (size_t)(m0 + ln) * K;
    const u16* aRow1 = A + (size_t)(m0 + 16 + ln) * K;

    // Each thread async-copies 2 x 16B chunks of the 128x32 B tile.
    const int chunk0 = threadIdx.x;           // chunk = row*4 + kc
    auto issueB = [&](int buf, int k0) {
        #pragma unroll
        for (int i = 0; i < 2; ++i) {
            const int chunk = chunk0 + i * 256;
            const int row = chunk >> 2;       // 0..127 (block-local n)
            const int kc  = chunk & 3;        // 16B sub-chunk of the 32-K row
            const u16* src = BT + (size_t)(nBlock + row) * K + k0 + kc * 8;
            uint32_t laddr = (uint32_t)(uintptr_t)&bsmem[buf][row * BSTRIDE + kc * 8];
            async_g2l_b128(laddr, src);
        }
    };

    issueB(0, 0);
    int buf = 0;

    v8f c[2][4] = {};
    for (int k0 = 0; k0 < K; k0 += 32) {
        wait_async0();          // own async issues complete
        __syncthreads();        // -> whole B tile complete for every wave
        if (k0 + 32 < K)
            issueB(buf ^ 1, k0 + 32);   // overlap next fill with this compute
        if (k0 + 128 < K)
            __builtin_prefetch(aRow0 + k0 + 128, 0, 1);

        Frag16 a0, a1, b0, b1, b2, b3;
        loadA(a0, aRow0, k0, half);
        loadA(a1, aRow1, k0, half);
        const u16* bl = &bsmem[buf][0];
        loadB_lds(b0, bl + (wn * 64 +  0 + ln) * BSTRIDE, half);
        loadB_lds(b1, bl + (wn * 64 + 16 + ln) * BSTRIDE, half);
        loadB_lds(b2, bl + (wn * 64 + 32 + ln) * BSTRIDE, half);
        loadB_lds(b3, bl + (wn * 64 + 48 + ln) * BSTRIDE, half);

        c[0][0] = wmma_bf16(a0, b0, c[0][0]);
        c[0][1] = wmma_bf16(a0, b1, c[0][1]);
        c[0][2] = wmma_bf16(a0, b2, c[0][2]);
        c[0][3] = wmma_bf16(a0, b3, c[0][3]);
        c[1][0] = wmma_bf16(a1, b0, c[1][0]);
        c[1][1] = wmma_bf16(a1, b1, c[1][1]);
        c[1][2] = wmma_bf16(a1, b2, c[1][2]);
        c[1][3] = wmma_bf16(a1, b3, c[1][3]);

        __syncthreads();        // all waves done reading buf before it is refilled
        buf ^= 1;
    }

    #pragma unroll
    for (int ni = 0; ni < 4; ++ni) {
        const float bv = bias ? bias[n0 + ni * 16 + ln] : 0.0f;
        #pragma unroll
        for (int mi = 0; mi < 2; ++mi) {
            if (outF32) {
                float* o = (float*)out;
                #pragma unroll
                for (int r = 0; r < 8; ++r)
                    o[(size_t)(m0 + mi * 16 + half * 8 + r) * N + n0 + ni * 16 + ln] =
                        c[mi][ni][r] + bv;
            } else {
                u16* o = (u16*)out;
                #pragma unroll
                for (int r = 0; r < 8; ++r)
                    o[(size_t)(m0 + mi * 16 + half * 8 + r) * N + n0 + ni * 16 + ln] =
                        f2bf(c[mi][ni][r] + bv);
            }
        }
    }
}

// ---------------------------------------------------------------------------
// Softmax of q over the sequence axis (column softmax over 4096 rows), *SCALE.
// One 256-thread block per (b, c), c = h*64+d. Writes bf16 transposed
// qT[b][c][s] (contiguous in s -> contiguous WMMA operand later).
// ---------------------------------------------------------------------------
__global__ __launch_bounds__(256) void softmax_seq(
    const u16* __restrict__ qkv, u16* __restrict__ qT)
{
    const int c = blockIdx.x;
    const int b = blockIdx.y;
    const int t = threadIdx.x;
    const int lane = t & 31, w = t >> 5;
    __shared__ float redA[8], redB[8];

    const u16* base = qkv + (size_t)b * S_ * (3 * DIM_) + c;
    float v[16];
    #pragma unroll
    for (int i = 0; i < 16; ++i)
        v[i] = bf2f(base[(size_t)(t + i * 256) * (3 * DIM_)]);

    float m = -3.4e38f;
    #pragma unroll
    for (int i = 0; i < 16; ++i) m = fmaxf(m, v[i]);
    #pragma unroll
    for (int off = 16; off > 0; off >>= 1) m = fmaxf(m, __shfl_xor(m, off));
    if (lane == 0) redA[w] = m;
    __syncthreads();
    float gm = redA[0];
    #pragma unroll
    for (int i = 1; i < 8; ++i) gm = fmaxf(gm, redA[i]);

    float s = 0.0f;
    #pragma unroll
    for (int i = 0; i < 16; ++i) { v[i] = __expf(v[i] - gm); s += v[i]; }
    #pragma unroll
    for (int off = 16; off > 0; off >>= 1) s += __shfl_xor(s, off);
    if (lane == 0) redB[w] = s;
    __syncthreads();
    float gs = 0.0f;
    #pragma unroll
    for (int i = 0; i < 8; ++i) gs += redB[i];

    const float inv = SCALE_ / gs;
    u16* o = qT + ((size_t)b * DIM_ + c) * (size_t)S_;
    #pragma unroll
    for (int i = 0; i < 16; ++i) o[t + i * 256] = f2bf(v[i] * inv);
}

// ---------------------------------------------------------------------------
// Softmax of k over head_dim (row softmax over 64). One wave per (b,s,h) row.
// Writes bf16 transposed: kT[b][h*64+d][s].
// ---------------------------------------------------------------------------
__global__ __launch_bounds__(256) void softmax_hd(
    const u16* __restrict__ qkv, u16* __restrict__ kT)
{
    const int lane = threadIdx.x & 31;
    const size_t gw = (size_t)blockIdx.x * 8 + (threadIdx.x >> 5); // (b*S+s)*H+h
    const int h  = (int)(gw & (H_ - 1));
    const size_t bs = gw >> 4;              // b*S + s
    const int b = (int)(bs >> 12);
    const int s = (int)(bs & (S_ - 1));

    const u16* row = qkv + bs * (3 * DIM_) + DIM_ + h * HD_;
    float v0 = bf2f(row[lane]);
    float v1 = bf2f(row[lane + 32]);
    float m = fmaxf(v0, v1);
    #pragma unroll
    for (int off = 16; off > 0; off >>= 1) m = fmaxf(m, __shfl_xor(m, off));
    v0 = __expf(v0 - m);
    v1 = __expf(v1 - m);
    float sum = v0 + v1;
    #pragma unroll
    for (int off = 16; off > 0; off >>= 1) sum += __shfl_xor(sum, off);
    const float inv = 1.0f / sum;

    u16* o = kT + ((size_t)b * DIM_ + h * HD_) * (size_t)S_ + s;
    o[(size_t)lane * S_]        = f2bf(v0 * inv);
    o[(size_t)(lane + 32) * S_] = f2bf(v1 * inv);
}

// ---------------------------------------------------------------------------
// A[b,h,d,e] = sum_s k[s,d] * q[s,e], K split into 4 chunks of 1024 for
// parallelism (8192 waves). f32 partials in At layout: P[chunk][bh][e][d].
// ---------------------------------------------------------------------------
__global__ __launch_bounds__(256) void kq_gemm_split(
    const u16* __restrict__ kT, const u16* __restrict__ qT, float* __restrict__ P)
{
    const int lane = threadIdx.x & 31, w = threadIdx.x >> 5;
    const int half = lane >> 4, ln = lane & 15;
    const int gid = blockIdx.x * 8 + w;
    const int chunk = gid & 3;
    const int tile  = (gid >> 2) & 15;
    const int bh    = gid >> 6;
    const int d0 = (tile & 3) * 16, e0 = (tile >> 2) * 16;
    const int kbeg = chunk * (S_ / 4), kend = kbeg + S_ / 4;

    const u16* aRow = kT + ((size_t)bh * HD_ + d0 + ln) * (size_t)S_;
    const u16* bRow = qT + ((size_t)bh * HD_ + e0 + ln) * (size_t)S_;

    v8f c = {};
    for (int k0 = kbeg; k0 < kend; k0 += 32) {
        Frag16 a, b;
        loadA(a, aRow, k0, half);
        loadB(b, bRow, k0, half);
        c = wmma_bf16(a, b, c);
    }
    float* o = P + ((size_t)chunk * (B_ * H_) + bh) * (HD_ * HD_)
                 + (size_t)(e0 + ln) * HD_ + d0 + half * 8;
    #pragma unroll
    for (int r = 0; r < 8; ++r) o[r] = c[r];
}

// Sum the 4 K-chunk partials -> At[e][d] bf16 (B-operand of v@A).
__global__ __launch_bounds__(256) void reduce_at(
    const float* __restrict__ P, u16* __restrict__ At, size_t n)
{
    size_t i = (size_t)blockIdx.x * 256 + threadIdx.x;
    if (i < n)
        At[i] = f2bf(P[i] + P[i + n] + P[i + 2 * n] + P[i + 3 * n]);
}

// ---------------------------------------------------------------------------
// out[b,h,s,e] = sum_d v[s,d] * A[d,e].  v read straight out of the qkv buffer
// (rows contiguous in d); At rows are the contiguous B-operand. Result packed
// into attn[b*S+s][h*64+e] bf16 for the projection GEMM.
// ---------------------------------------------------------------------------
__global__ __launch_bounds__(256) void va_gemm(
    const u16* __restrict__ qkv, const u16* __restrict__ At, u16* __restrict__ attn)
{
    const int lane = threadIdx.x & 31, w = threadIdx.x >> 5;
    const int half = lane >> 4, ln = lane & 15;
    size_t gt = (size_t)blockIdx.x * 8 + w;
    const int e0 = (int)(gt & 3) * 16;
    gt >>= 2;
    const int s0 = (int)(gt & 255) * 16;
    const int bh = (int)(gt >> 8);
    const int h = bh & (H_ - 1), b = bh >> 4;

    const u16* aRow = qkv + (size_t)(b * S_ + s0 + ln) * (3 * DIM_) + 2 * DIM_ + h * HD_;
    const u16* bRow = At + (size_t)bh * (HD_ * HD_) + (size_t)(e0 + ln) * HD_;

    v8f c = {};
    #pragma unroll
    for (int k0 = 0; k0 < HD_; k0 += 32) {
        Frag16 a, bb;
        loadA(a, aRow, k0, half);
        loadB(bb, bRow, k0, half);
        c = wmma_bf16(a, bb, c);
    }
    u16* o = attn + (size_t)(b * S_ + s0 + half * 8) * DIM_ + h * HD_ + e0 + ln;
    #pragma unroll
    for (int r = 0; r < 8; ++r) o[(size_t)r * DIM_] = f2bf(c[r]);
}

// ---------------------------------------------------------------------------
// Prep: f32 -> bf16 convert, and f32 [K,N] -> bf16 [N,K] transpose (weights).
// ---------------------------------------------------------------------------
__global__ __launch_bounds__(256) void cvt_f32_bf16(
    const float* __restrict__ in, u16* __restrict__ out, size_t n)
{
    size_t i = (size_t)blockIdx.x * 256 + threadIdx.x;
    if (i < n) out[i] = f2bf(in[i]);
}

__global__ __launch_bounds__(256) void transpose_f32_bf16(
    const float* __restrict__ in, u16* __restrict__ out, int K, int N)
{
    size_t i = (size_t)blockIdx.x * 256 + threadIdx.x;
    if (i < (size_t)K * N) {
        int k = (int)(i / N), n = (int)(i % N);
        out[(size_t)n * K + k] = f2bf(in[i]);
    }
}

// ---------------------------------------------------------------------------
// Launch: x, Wqkv, bqkv, Wproj, bproj  ->  d_out f32 [B,S,DIM]
// ---------------------------------------------------------------------------
extern "C" void kernel_launch(void* const* d_in, const int* in_sizes, int n_in,
                              void* d_out, int out_size, void* d_ws, size_t ws_size,
                              hipStream_t stream)
{
    (void)in_sizes; (void)n_in; (void)out_size; (void)ws_size;
    const float* x     = (const float*)d_in[0];
    const float* Wqkv  = (const float*)d_in[1];
    const float* bqkv  = (const float*)d_in[2];
    const float* Wproj = (const float*)d_in[3];
    const float* bproj = (const float*)d_in[4];
    float* out = (float*)d_out;

    char* ws = (char*)d_ws;
    size_t off = 0;
    auto take = [&](size_t bytes) -> char* {
        char* p = ws + off;
        off = (off + bytes + 255) & ~(size_t)255;
        return p;
    };

    const size_t MS = (size_t)B_ * S_;                    // 32768 rows
    u16*   xbf    = (u16*)take(MS * DIM_ * 2);              //  64 MiB
    u16*   WqkvT  = (u16*)take((size_t)3 * DIM_ * DIM_ * 2);//   6 MiB
    u16*   WprojT = (u16*)take((size_t)DIM_ * DIM_ * 2);    //   2 MiB
    u16*   qkv    = (u16*)take(MS * 3 * DIM_ * 2);          // 192 MiB
    u16*   qT     = (u16*)take(MS * DIM_ * 2);              //  64 MiB
    u16*   kT     = (u16*)take(MS * DIM_ * 2);              //  64 MiB
    u16*   At     = (u16*)take((size_t)B_ * H_ * HD_ * HD_ * 2); // 1 MiB
    float* Ap     = (float*)take((size_t)4 * B_ * H_ * HD_ * HD_ * 4); // 8 MiB partials
    u16*   attn   = (u16*)take(MS * DIM_ * 2);              //  64 MiB

    // 1. x -> bf16
    {
        size_t n = MS * DIM_;
        cvt_f32_bf16<<<dim3((unsigned)((n + 255) / 256)), dim3(256), 0, stream>>>(x, xbf, n);
    }
    // 2-3. weight transposes -> bf16 [N,K]
    transpose_f32_bf16<<<dim3((3u * DIM_ * DIM_ + 255) / 256), dim3(256), 0, stream>>>(
        Wqkv, WqkvT, DIM_, 3 * DIM_);
    transpose_f32_bf16<<<dim3((DIM_ * DIM_ + 255) / 256), dim3(256), 0, stream>>>(
        Wproj, WprojT, DIM_, DIM_);
    // 4. qkv = x @ Wqkv + bqkv   (M=32768, N=3072, K=1024) -> bf16
    gemm_bf16_wmma<<<dim3(3 * DIM_ / 128, (unsigned)(MS / 128)), dim3(256), 0, stream>>>(
        xbf, WqkvT, bqkv, qkv, 3 * DIM_, DIM_, 0);
    // 5. q: softmax over seqlen, * SCALE -> qT[b][c][s]
    softmax_seq<<<dim3(DIM_, B_), dim3(256), 0, stream>>>(qkv, qT);
    // 6. k: softmax over head_dim -> kT[b][c][s]
    softmax_hd<<<dim3((unsigned)(MS * H_ / 8)), dim3(256), 0, stream>>>(qkv, kT);
    // 7. A = k^T q per (b,h), K split 4 ways -> f32 partials
    kq_gemm_split<<<dim3(B_ * H_ * 16 * 4 / 8), dim3(256), 0, stream>>>(kT, qT, Ap);
    //    reduce partials -> At[e][d] bf16
    {
        size_t n = (size_t)B_ * H_ * HD_ * HD_;   // 524288
        reduce_at<<<dim3((unsigned)((n + 255) / 256)), dim3(256), 0, stream>>>(Ap, At, n);
    }
    // 8. attn = v @ A
    va_gemm<<<dim3((unsigned)(B_ * H_ * (S_ / 16) * 4 / 8)), dim3(256), 0, stream>>>(
        qkv, At, attn);
    // 9. out = attn @ Wproj + bproj  (f32 to d_out)
    gemm_bf16_wmma<<<dim3(DIM_ / 128, (unsigned)(MS / 128)), dim3(256), 0, stream>>>(
        attn, WprojT, bproj, out, DIM_, DIM_, 1);
}
